// LSTM_1_76647986364858
// MI455X (gfx1250) — compile-verified, matrix operand order checked by reading, not run on hardware
//
#include <hip/hip_runtime.h>
#include <hip/hip_bf16.h>

// ---------------------------------------------------------------------------
// Multi-task LSTM scan for MI455X (gfx1250, wave32, WMMA).
//
// x[B=512][S=256][I=64], H=64, T=4 task cells. Output [T][B][S][H] f32.
// Persistent-tile design: each workgroup owns a 16-row batch tile (WMMA M=16)
// and runs the full 256-step recurrence with 32 waves (1024 threads, 8
// waves/SIMD). All GEMMs are v_wmma_f32_16x16x32_f16 with f32 accumulation.
//
// Weights are converted once to f16 into workspace by a prep kernel (~504KB,
// L2-resident); B-fragments are two 16B global loads, no per-step cvt.
// Gate GEMMs are split across the full 32-wave complement (K-split for the
// shared cell, tile-split for the task cells): ~21 serially-dependent WMMAs
// per step. The x tile is software-pipelined one step ahead in registers so
// its HBM latency hides behind the previous step's phases.
// ---------------------------------------------------------------------------

typedef __attribute__((ext_vector_type(16))) _Float16 v16h;
typedef __attribute__((ext_vector_type(8)))  _Float16 v8h;
typedef __attribute__((ext_vector_type(4)))  _Float16 v4h;
typedef __attribute__((ext_vector_type(8)))  float    v8f;
typedef __attribute__((ext_vector_type(4)))  float    v4f;

struct KParams {
  const float* x;
  const float* WxT[4]; const float* WhT[4]; const float* bT[4];  // task i,f,g,o
  const float* WxS[4]; const float* WhS[4]; const float* bS[4];  // shared
  const float* Wmv; const float* bmv;
  const float* Wfa; const float* bfa;
  const float* Wsa; const float* bsa;
  const _Float16* W16;
  float* out;
};

// f16 weight pack layout (offsets in halfs)
#define OFF_MV   0
#define OFF_FA   4096
#define OFF_SA   8192
#define OFF_SWX  12288    // 4 x [64][128]  (+g*8192)
#define OFF_SWH  45056    // 4 x [64][64]   (+g*4096)
#define OFF_TWX  61440    // 4 x [4][64][128] (+g*32768 + t*8192)
#define OFF_TWH  192512   // 4 x [4][64][64]  (+g*16384 + t*4096)
#define W16_TOTAL 258048
#define W16_BYTES (W16_TOTAL * 2)

#define NT 1024   // threads per block (32 waves)

__device__ __forceinline__ float sigm(float x) { return 1.0f / (1.0f + __expf(-x)); }

__device__ __forceinline__ v8f wmma_f16(v16h a, v16h b, v8f c) {
  return __builtin_amdgcn_wmma_f32_16x16x32_f16(false, a, false, b, (short)0, c, false, false);
}

// A-fragment: 16x32 f16 from LDS row-major [M][stride halfs].
__device__ __forceinline__ v16h fragA_lds(const _Float16* base, int stride, int kc, int lane) {
  int m  = lane & 15;
  int ko = kc * 32 + ((lane >> 4) << 3);
  const _Float16* p = base + m * stride + ko;
  v8h lo = *(const v8h*)(p);
  v8h hi = *(const v8h*)(p + 16);
  v16h r;
#pragma unroll
  for (int j = 0; j < 8; ++j) { r[j] = lo[j]; r[j + 8] = hi[j]; }
  return r;
}

// B-fragment from packed f16 weights, row-major [N][ldk halfs].
__device__ __forceinline__ v16h fragB16(const _Float16* W, int n0, int ldk, int kc, int lane) {
  int n  = n0 + (lane & 15);
  int ko = kc * 32 + ((lane >> 4) << 3);
  const _Float16* p = W + n * ldk + ko;
  v8h lo = *(const v8h*)(p);
  v8h hi = *(const v8h*)(p + 16);
  v16h r;
#pragma unroll
  for (int j = 0; j < 8; ++j) { r[j] = lo[j]; r[j + 8] = hi[j]; }
  return r;
}

// Fallback: B-fragment converted on the fly from f32 weights [N][ldk].
__device__ __forceinline__ v16h fragB_g32(const float* W, int n0, int ldk, int kc, int lane) {
  int n  = n0 + (lane & 15);
  int ko = kc * 32 + ((lane >> 4) << 3);
  const float* p = W + n * ldk + ko;
  v4f a0 = *(const v4f*)(p);
  v4f a1 = *(const v4f*)(p + 4);
  v4f b0 = *(const v4f*)(p + 16);
  v4f b1 = *(const v4f*)(p + 20);
  v16h r;
#pragma unroll
  for (int j = 0; j < 4; ++j) {
    r[j]      = (_Float16)a0[j];
    r[4 + j]  = (_Float16)a1[j];
    r[8 + j]  = (_Float16)b0[j];
    r[12 + j] = (_Float16)b1[j];
  }
  return r;
}

template <bool F16W>
__device__ __forceinline__ v16h loadB(const float* Wf, const _Float16* Wh16,
                                      int n0, int ldk, int kc, int lane) {
  if constexpr (F16W) return fragB16(Wh16, n0, ldk, kc, lane);
  else                return fragB_g32(Wf, n0, ldk, kc, lane);
}

// ---- prep kernel: convert all weights f32 -> f16 into ws ----
struct PrepArgs { const float* src[19]; int cnt[19]; };

__global__ __launch_bounds__(256) void convert_weights(PrepArgs a, _Float16* dst, int total) {
  int idx = blockIdx.x * blockDim.x + threadIdx.x;
  if (idx >= total) return;
  int rem = idx;
#pragma unroll
  for (int sgi = 0; sgi < 19; ++sgi) {
    if (rem < a.cnt[sgi]) { dst[idx] = (_Float16)a.src[sgi][rem]; return; }
    rem -= a.cnt[sgi];
  }
}

#define LA_S 200   // shared-cell A [x_att | h_s | h_s]
#define LA_T 136   // task common A [x_att | s_att]
#define LA_H 72    // per-cell h_t
#define LXS  72    // x / xm staging

template <bool F16W>
__global__ __launch_bounds__(NT, 1) void mtlstm_kernel(KParams p, int S, int Bfull) {
  __shared__ __align__(16) _Float16 sAS[16 * LA_S];
  __shared__ __align__(16) _Float16 sAT[16 * LA_T];
  __shared__ __align__(16) _Float16 sAH[4 * 16 * LA_H];
  __shared__ __align__(16) _Float16 sXin[16 * LXS];
  __shared__ __align__(16) _Float16 sXm[16 * LXS];
  __shared__ float sGpre[2 * 4 * 16 * 64];   // shared-cell gate partials [kh][g][M][n]
  __shared__ float sTpre[4 * 4 * 16 * 64];   // task gate pre-acts [g][t][M][n]
  __shared__ float sCS[16 * 64];
  __shared__ float sCT[4 * 16 * 64];
  __shared__ float sB[23 * 64];

  const int tid  = threadIdx.x;
  const int wave = tid >> 5;
  const int lane = tid & 31;
  const int m0   = blockIdx.x * 16;

  // ---- one-time init ----
  for (int e = tid; e < 23 * 64; e += NT) {
    float v;
    if (e < 64)        v = p.bmv[e];
    else if (e < 128)  v = p.bfa[e - 64];
    else if (e < 192)  v = p.bsa[e - 128];
    else if (e < 448)  { int r = e - 192; v = p.bS[r >> 6][r & 63]; }
    else               { int r = e - 448; v = p.bT[r >> 8][r & 255]; }
    sB[e] = v;
  }
  for (int e = tid; e < 16 * LA_S; e += NT) sAS[e] = (_Float16)0.0f;
  for (int e = tid; e < 16 * LA_T; e += NT) sAT[e] = (_Float16)0.0f;
  for (int e = tid; e < 4 * 16 * LA_H; e += NT) sAH[e] = (_Float16)0.0f;
  for (int e = tid; e < 16 * LXS; e += NT) { sXin[e] = (_Float16)0.0f; sXm[e] = (_Float16)0.0f; }
  for (int e = tid; e < 16 * 64; e += NT) sCS[e] = 0.0f;
  for (int e = tid; e < 4 * 16 * 64; e += NT) sCT[e] = 0.0f;
  __syncthreads();

  const int N   = lane & 15;
  const int hi8 = (lane >> 4) << 3;
  const _Float16* W16 = p.W16;

  // x software pipeline: threads 0-255 each own one float4 of the tile.
  const int xm_ = tid >> 4, xi4 = (tid & 15) * 4;
  const float* xptr = p.x + ((size_t)(m0 + xm_) * S) * 64 + xi4;
  v4f xv = {};
  if (tid < 256) xv = *(const v4f*)(xptr);              // prefetch step 0

#pragma unroll 1
  for (int s = 0; s < S; ++s) {
    // ---- commit prefetched x_t to LDS (f32x4 -> f16x4) ----
    if (tid < 256) {
      v4h hv; hv[0] = (_Float16)xv[0]; hv[1] = (_Float16)xv[1];
      hv[2] = (_Float16)xv[2]; hv[3] = (_Float16)xv[3];
      *(v4h*)(&sXin[xm_ * LXS + xi4]) = hv;
    }
    __syncthreads();
    // issue next step's x load; consumed at next loop top (latency hidden
    // behind this step's phases)
    if (tid < 256 && (s + 1) < S) xv = *(const v4f*)(xptr + (size_t)(s + 1) * 64);

    // ---- Phase A: xm = x @ Wmv.T + bmv (waves 0-3) ----
    if (wave < 4) {
      int nt = wave;
      v8f acc = {};
#pragma unroll
      for (int kc = 0; kc < 2; ++kc)
        acc = wmma_f16(fragA_lds(sXin, LXS, kc, lane),
                       loadB<F16W>(p.Wmv, W16 + OFF_MV, nt * 16, 64, kc, lane), acc);
      int n = nt * 16 + N;
#pragma unroll
      for (int v = 0; v < 8; ++v) {
        int M = v + hi8;
        sXm[M * LXS + n] = (_Float16)(acc[v] + sB[n]);
      }
    }
    __syncthreads();

    // ---- Phase B: x_att = sigmoid(xm @ Wfa.T + bfa) * xm (waves 0-3) ----
    if (wave < 4) {
      int nt = wave;
      v8f acc = {};
#pragma unroll
      for (int kc = 0; kc < 2; ++kc)
        acc = wmma_f16(fragA_lds(sXm, LXS, kc, lane),
                       loadB<F16W>(p.Wfa, W16 + OFF_FA, nt * 16, 64, kc, lane), acc);
      int n = nt * 16 + N;
#pragma unroll
      for (int v = 0; v < 8; ++v) {
        int M = v + hi8;
        float xa = sigm(acc[v] + sB[64 + n]) * (float)sXm[M * LXS + n];
        _Float16 h16 = (_Float16)xa;
        sAS[M * LA_S + n] = h16;
        sAT[M * LA_T + n] = h16;
      }
    }
    __syncthreads();

    // ---- Phase C: shared-cell gates, K-split across wave pairs.
    //      unit u = (g,nt), kh = K half; 3 WMMAs per wave. ----
    {
      int u  = wave >> 1;
      int g  = u >> 2;
      int nt = u & 3;
      int kh = wave & 1;
      v8f acc = {};
#pragma unroll
      for (int j = 0; j < 3; ++j) {
        int kc = kh * 3 + j;
        v16h a = fragA_lds(sAS, LA_S, kc, lane);
        v16h b = (kc < 4)
          ? loadB<F16W>(p.WxS[g], W16 + OFF_SWX + g * 8192, nt * 16, 128, kc, lane)
          : loadB<F16W>(p.WhS[g], W16 + OFF_SWH + g * 4096, nt * 16, 64, kc - 4, lane);
        acc = wmma_f16(a, b, acc);
      }
      int n = nt * 16 + N;
#pragma unroll
      for (int v = 0; v < 8; ++v) {
        int M = v + hi8;
        sGpre[kh * 4096 + g * 1024 + M * 64 + n] = acc[v];
      }
    }
    __syncthreads();

    // ---- Phase C2: shared-cell elementwise c,h (one element per thread) ----
    {
      int e = tid;  // 16*64 = 1024 elements
      int m = e >> 6, n = e & 63;
      float pi = sGpre[e]        + sGpre[4096 + e]        + sB[192 + n];
      float pf = sGpre[1024 + e] + sGpre[4096 + 1024 + e] + sB[256 + n];
      float pg = sGpre[2048 + e] + sGpre[4096 + 2048 + e] + sB[320 + n];
      float po = sGpre[3072 + e] + sGpre[4096 + 3072 + e] + sB[384 + n];
      float gi = sigm(pi), gf = sigm(pf), gg = tanhf(pg), go = sigm(po);
      float c = gf * sCS[e] + gi * gg;
      float h = go * tanhf(c);
      sCS[e] = c;
      _Float16 h16 = (_Float16)h;
      sAS[m * LA_S + 64 + n]  = h16;
      sAS[m * LA_S + 128 + n] = h16;
    }
    __syncthreads();

    // ---- Phase D: s_att = sigmoid(h_s @ Wsa.T + bsa) * h_s (waves 0-3) ----
    if (wave < 4) {
      int nt = wave;
      v8f acc = {};
#pragma unroll
      for (int kc = 0; kc < 2; ++kc)
        acc = wmma_f16(fragA_lds(sAS + 64, LA_S, kc, lane),
                       loadB<F16W>(p.Wsa, W16 + OFF_SA, nt * 16, 64, kc, lane), acc);
      int n = nt * 16 + N;
#pragma unroll
      for (int v = 0; v < 8; ++v) {
        int M = v + hi8;
        float sa = sigm(acc[v] + sB[128 + n]) * (float)sAS[M * LA_S + 64 + n];
        sAT[M * LA_T + 64 + n] = (_Float16)sa;
      }
    }
    __syncthreads();

    // ---- Phase E: task-cell gates. 64 (t,nt,g) tiles over 32 waves:
    //      each wave does gates {g0,g0+1} of one (t,nt), sharing A-fragments;
    //      12 WMMAs per wave. Pre-acts staged to LDS. ----
    {
      int t  = wave >> 3;
      int nt = (wave >> 1) & 3;
      int g0 = (wave & 1) * 2;
      const _Float16* aH = sAH + t * 16 * LA_H;
      const float* Wxf0 = p.WxT[g0]     + t * 8192;
      const float* Wxf1 = p.WxT[g0 + 1] + t * 8192;
      const float* Whf0 = p.WhT[g0]     + t * 4096;
      const float* Whf1 = p.WhT[g0 + 1] + t * 4096;
      const _Float16* Wxh0 = W16 + OFF_TWX + g0 * 32768       + t * 8192;
      const _Float16* Wxh1 = W16 + OFF_TWX + (g0 + 1) * 32768 + t * 8192;
      const _Float16* Whh0 = W16 + OFF_TWH + g0 * 16384       + t * 4096;
      const _Float16* Whh1 = W16 + OFF_TWH + (g0 + 1) * 16384 + t * 4096;
      v8f acc0 = {}, acc1 = {};
#pragma unroll
      for (int kc = 0; kc < 6; ++kc) {
        v16h a = (kc < 4) ? fragA_lds(sAT, LA_T, kc, lane)
                          : fragA_lds(aH, LA_H, kc - 4, lane);
        v16h b0 = (kc < 4) ? loadB<F16W>(Wxf0, Wxh0, nt * 16, 128, kc, lane)
                           : loadB<F16W>(Whf0, Whh0, nt * 16, 64, kc - 4, lane);
        acc0 = wmma_f16(a, b0, acc0);
        v16h b1 = (kc < 4) ? loadB<F16W>(Wxf1, Wxh1, nt * 16, 128, kc, lane)
                           : loadB<F16W>(Whf1, Whh1, nt * 16, 64, kc - 4, lane);
        acc1 = wmma_f16(a, b1, acc1);
      }
      int n = nt * 16 + N;
#pragma unroll
      for (int v = 0; v < 8; ++v) {
        int M = v + hi8;
        sTpre[g0 * 4096       + t * 1024 + M * 64 + n] = acc0[v];
        sTpre[(g0 + 1) * 4096 + t * 1024 + M * 64 + n] = acc1[v];
      }
    }
    __syncthreads();

    // ---- Phase E2: task-cell elementwise c,h + output store (4 per thread) ----
    for (int e = tid; e < 4096; e += NT) {
      int t = e >> 10, m = (e >> 6) & 15, n = e & 63;
      float gi = sigm(sTpre[e]          + sB[448 + 0 * 256 + t * 64 + n]);
      float gf = sigm(sTpre[4096 + e]   + sB[448 + 1 * 256 + t * 64 + n]);
      float gg = tanhf(sTpre[8192 + e]  + sB[448 + 2 * 256 + t * 64 + n]);
      float go = sigm(sTpre[12288 + e]  + sB[448 + 3 * 256 + t * 64 + n]);
      float c = gf * sCT[e] + gi * gg;
      float h = go * tanhf(c);
      sCT[e] = c;
      sAH[t * 16 * LA_H + m * LA_H + n] = (_Float16)h;
      p.out[(((size_t)t * Bfull + (m0 + m)) * S + s) * 64 + n] = h;
    }
    __syncthreads();
  }
}

extern "C" void kernel_launch(void* const* d_in, const int* in_sizes, int n_in,
                              void* d_out, int out_size, void* d_ws, size_t ws_size,
                              hipStream_t stream) {
  (void)n_in; (void)out_size;

  // setup_inputs() insertion order:
  //  0: x; 1..12: task {Wxi,bi,Whi,Wxf,bf,Whf,Wxg,bg,Whg,Wxo,bo,Who} (lead T=4)
  // 13..24: shared {same}; 25..30: W_mv,b_mv,W_fa,b_fa,W_sa,b_sa
  KParams kp;
  kp.x = (const float*)d_in[0];
  const int wxT[4] = {1, 4, 7, 10}, bTi[4] = {2, 5, 8, 11}, whT[4] = {3, 6, 9, 12};
  const int wxS[4] = {13, 16, 19, 22}, bSi[4] = {14, 17, 20, 23}, whS[4] = {15, 18, 21, 24};
  for (int g = 0; g < 4; ++g) {
    kp.WxT[g] = (const float*)d_in[wxT[g]];
    kp.bT[g]  = (const float*)d_in[bTi[g]];
    kp.WhT[g] = (const float*)d_in[whT[g]];
    kp.WxS[g] = (const float*)d_in[wxS[g]];
    kp.bS[g]  = (const float*)d_in[bSi[g]];
    kp.WhS[g] = (const float*)d_in[whS[g]];
  }
  kp.Wmv = (const float*)d_in[25]; kp.bmv = (const float*)d_in[26];
  kp.Wfa = (const float*)d_in[27]; kp.bfa = (const float*)d_in[28];
  kp.Wsa = (const float*)d_in[29]; kp.bsa = (const float*)d_in[30];
  kp.out = (float*)d_out;
  kp.W16 = (const _Float16*)d_ws;

  const int S = 256;
  const int Bfull = in_sizes[0] / (S * 64);   // 512
  dim3 grid(Bfull / 16), block(NT);

  const bool useF16W = (ws_size >= (size_t)W16_BYTES) && (d_ws != nullptr);
  if (useF16W) {
    PrepArgs pa;
    int si = 0;
    pa.src[si] = kp.Wmv; pa.cnt[si++] = 4096;
    pa.src[si] = kp.Wfa; pa.cnt[si++] = 4096;
    pa.src[si] = kp.Wsa; pa.cnt[si++] = 4096;
    for (int g = 0; g < 4; ++g) { pa.src[si] = kp.WxS[g]; pa.cnt[si++] = 8192; }
    for (int g = 0; g < 4; ++g) { pa.src[si] = kp.WhS[g]; pa.cnt[si++] = 4096; }
    for (int g = 0; g < 4; ++g) { pa.src[si] = kp.WxT[g]; pa.cnt[si++] = 32768; }
    for (int g = 0; g < 4; ++g) { pa.src[si] = kp.WhT[g]; pa.cnt[si++] = 16384; }
    convert_weights<<<dim3((W16_TOTAL + 255) / 256), dim3(256), 0, stream>>>(
        pa, (_Float16*)d_ws, W16_TOTAL);
    mtlstm_kernel<true><<<grid, block, 0, stream>>>(kp, S, Bfull);
  } else {
    mtlstm_kernel<false><<<grid, block, 0, stream>>>(kp, S, Bfull);
  }
}
